// StaticDotAttention_58626303590870
// MI455X (gfx1250) — matrix operand order repaired and checked by reading, not compiled, not run on hardware
//
#include <hip/hip_runtime.h>
#include <hip/hip_bf16.h>
#include <cstdint>
#include <cstddef>

// ---------------------------------------------------------------------------
// StaticDotAttention for MI455X (gfx1250, wave32, WMMA).
// out = concat( softmax_masked( relu(inp W^T) relu(mem W^T)^T / sqrt(A) ) @ mem , inp )
// B=32, Lq=Lk=1024, D=A=768.
// All four GEMMs run on v_wmma_f32_16x16x32_bf16 (fp32 accumulate).
// Intermediates (inp_p, mem_p, probs: ~168 MB bf16) live in workspace and fit
// in the 192 MB L2, so cross-kernel operand traffic is L2-resident.
// ---------------------------------------------------------------------------

typedef __attribute__((ext_vector_type(16))) __bf16 v16bf;
typedef __attribute__((ext_vector_type(8)))  __bf16 v8bf;
typedef __attribute__((ext_vector_type(2)))  __bf16 v2bf;
typedef __attribute__((ext_vector_type(8)))  float  v8f;

static constexpr int kB  = 32;
static constexpr int kLq = 1024;
static constexpr int kLk = 1024;
static constexpr int kD  = 768;   // model dim
static constexpr int kA  = 768;   // projection dim

union ABFrag { v16bf v; v8bf h[2]; };

// Truncating fp32 -> bf16 (<=1 ulp vs RNE; inputs are fp32 noise-scale data).
__device__ __forceinline__ unsigned short f2bf(float f) {
  return (unsigned short)(__builtin_bit_cast(unsigned, f) >> 16);
}

#if defined(__has_builtin)
#if __has_builtin(__builtin_amdgcn_cvt_pk_bf16_f32)
#define HAVE_PK_BF16 1
#endif
#endif
#ifndef HAVE_PK_BF16
#define HAVE_PK_BF16 0
#endif

// Pack two floats into one DWORD of bf16 (lo, hi) -- single v_perm_b32 on the
// fallback path: byte sel 0x07060302 takes {hi[3],hi[2],lo[3],lo[2]}.
__device__ __forceinline__ unsigned cvt_pk_bf16(float lo, float hi) {
#if HAVE_PK_BF16
  v2bf r = __builtin_amdgcn_cvt_pk_bf16_f32(lo, hi);
  return __builtin_bit_cast(unsigned, r);
#else
  return __builtin_amdgcn_perm(__builtin_bit_cast(unsigned, hi),
                               __builtin_bit_cast(unsigned, lo),
                               0x07060302u);
#endif
}

// ---------------------------------------------------------------------------
// Kernel 1: Out[m, a] = bf16( relu( sum_d X[m, d] * W[a, d] ) )
// NT GEMM, M x 768, K = 768.  Block tile 128x128, 8 waves (2x4), wave 64x32.
// Double-buffered LDS (ping-pong) -> one barrier per K-step.
// ---------------------------------------------------------------------------
__device__ __forceinline__ void load_proj_tile(
    const float* __restrict__ X, const float* __restrict__ W,
    unsigned short (*Xs)[40], unsigned short (*Ws)[40],
    int m0, int n0, int kb, int ldr, int ldc)
{
  #pragma unroll
  for (int rr = 0; rr < 128; rr += 32) {
    float4 xv = *(const float4*)(X + (size_t)(m0 + rr + ldr) * kD + kb + ldc);
    *(unsigned*)&Xs[rr + ldr][ldc]     = cvt_pk_bf16(xv.x, xv.y);
    *(unsigned*)&Xs[rr + ldr][ldc + 2] = cvt_pk_bf16(xv.z, xv.w);
    float4 wv = *(const float4*)(W + (size_t)(n0 + rr + ldr) * kD + kb + ldc);
    *(unsigned*)&Ws[rr + ldr][ldc]     = cvt_pk_bf16(wv.x, wv.y);
    *(unsigned*)&Ws[rr + ldr][ldc + 2] = cvt_pk_bf16(wv.z, wv.w);
  }
}

__global__ __launch_bounds__(256) void proj_relu_kernel(
    const float* __restrict__ X, const float* __restrict__ W,
    unsigned short* __restrict__ Out)
{
  __shared__ unsigned short Xs[2][128][40];   // [m][k], pitch 40 (80B): conflict-free
  __shared__ unsigned short Ws[2][128][40];   // [a][k]

  const int tid  = threadIdx.x;
  const int m0   = blockIdx.x * 128;
  const int n0   = blockIdx.y * 128;
  const int wave = tid >> 5, lane = tid & 31;
  const int hf   = lane >> 4, l16 = lane & 15;
  const int wm   = wave & 1, wn = wave >> 1;     // wave grid 2 (M) x 4 (N)
  const int ldr  = tid >> 3;                      // loader row 0..31
  const int ldc  = (tid & 7) * 4;                 // loader col (float4)

  v8f acc[4][2] = {};

  load_proj_tile(X, W, Xs[0], Ws[0], m0, n0, 0, ldr, ldc);
  __syncthreads();

  for (int kt = 0; kt < kD / 32; ++kt) {
    const int cur = kt & 1;
    if (kt + 1 < kD / 32)                          // prefetch next tile
      load_proj_tile(X, W, Xs[cur ^ 1], Ws[cur ^ 1], m0, n0, (kt + 1) * 32,
                     ldr, ldc);

    ABFrag a[4], bfr[2];
    #pragma unroll
    for (int mt = 0; mt < 4; ++mt) {              // A 16x32: half selects K octet
      const int row = wm * 64 + mt * 16 + l16;
      a[mt].h[0] = *(const v8bf*)&Xs[cur][row][hf * 8];
      a[mt].h[1] = *(const v8bf*)&Xs[cur][row][16 + hf * 8];
    }
    #pragma unroll
    for (int nt = 0; nt < 2; ++nt) {              // B 32x16: half selects K octet
      const int col = wn * 32 + nt * 16 + l16;
      bfr[nt].h[0] = *(const v8bf*)&Ws[cur][col][hf * 8];
      bfr[nt].h[1] = *(const v8bf*)&Ws[cur][col][16 + hf * 8];
    }
    #pragma unroll
    for (int mt = 0; mt < 4; ++mt)
      #pragma unroll
      for (int nt = 0; nt < 2; ++nt)
        acc[mt][nt] = __builtin_amdgcn_wmma_f32_16x16x32_bf16(
            false, a[mt].v, false, bfr[nt].v, (short)0, acc[mt][nt], false, false);

    __syncthreads();
  }

  #pragma unroll
  for (int mt = 0; mt < 4; ++mt)
    #pragma unroll
    for (int nt = 0; nt < 2; ++nt)
      #pragma unroll
      for (int r = 0; r < 8; ++r) {               // C: M = 8*half + r, N = l16
        const int row = m0 + wm * 64 + mt * 16 + hf * 8 + r;
        const int col = n0 + wn * 32 + nt * 16 + l16;
        Out[(size_t)row * kA + col] = f2bf(fmaxf(acc[mt][nt][r], 0.0f));
      }
}

// ---------------------------------------------------------------------------
// Kernel 2: logits = Qp Kp^T / sqrt(A); masked softmax; store normalized bf16
// probs.  Block = 32 q-rows x 1024 k-cols; wave owns 128 cols (2x8 tiles).
// Operands come straight from L2-resident bf16 (contiguous 16B/32B per lane).
// ---------------------------------------------------------------------------
__global__ __launch_bounds__(256) void scores_softmax_kernel(
    const unsigned short* __restrict__ Qp_all,
    const unsigned short* __restrict__ Kp_all,
    const int* __restrict__ mask,
    unsigned short* __restrict__ probs)
{
  __shared__ float red[8][32];
  __shared__ float stat[32];

  const int tid  = threadIdx.x;
  const int b    = blockIdx.y;
  const int q0   = blockIdx.x * 32;
  const int wave = tid >> 5, lane = tid & 31;
  const int hf   = lane >> 4, l16 = lane & 15;
  const int nb   = wave * 128;                    // this wave's column base

  const unsigned short* Qp = Qp_all + ((size_t)b * kLq + q0) * kA;
  const unsigned short* Kp = Kp_all + (size_t)b * kLk * kA;

  v8f acc[2][8] = {};
  for (int kt = 0; kt < kA / 32; ++kt) {
    const int kb = kt * 32;
    ABFrag a[2];
    #pragma unroll
    for (int mt = 0; mt < 2; ++mt) {
      const unsigned short* p = Qp + (size_t)(mt * 16 + l16) * kA + kb;
      a[mt].h[0] = *(const v8bf*)(p + hf * 8);
      a[mt].h[1] = *(const v8bf*)(p + 16 + hf * 8);
    }
    #pragma unroll
    for (int ni = 0; ni < 8; ++ni) {
      ABFrag bf2;                                 // B[k][n] = Kp[n][k] (NT)
      const unsigned short* p =
          Kp + (size_t)(nb + ni * 16 + l16) * kA + kb + hf * 8;
      bf2.h[0] = *(const v8bf*)p;
      bf2.h[1] = *(const v8bf*)(p + 16);
      #pragma unroll
      for (int mt = 0; mt < 2; ++mt)
        acc[mt][ni] = __builtin_amdgcn_wmma_f32_16x16x32_bf16(
            false, a[mt].v, false, bf2.v, (short)0, acc[mt][ni], false, false);
    }
  }

  const float scale = 0.03608439182435161f;       // 1/sqrt(768)
  float mk[8];
  #pragma unroll
  for (int ni = 0; ni < 8; ++ni)
    mk[ni] = (float)mask[(size_t)b * kLk + nb + ni * 16 + l16];

  #pragma unroll
  for (int mt = 0; mt < 2; ++mt)
    #pragma unroll
    for (int ni = 0; ni < 8; ++ni)
      #pragma unroll
      for (int r = 0; r < 8; ++r)
        acc[mt][ni][r] *= scale * mk[ni];         // s = logits * mask

  // --- row max (rows split across the two 16-lane halves) ---
  #pragma unroll
  for (int mt = 0; mt < 2; ++mt)
    #pragma unroll
    for (int r = 0; r < 8; ++r) {
      float m = -3.0e38f;
      #pragma unroll
      for (int ni = 0; ni < 8; ++ni) m = fmaxf(m, acc[mt][ni][r]);
      #pragma unroll
      for (int off = 8; off >= 1; off >>= 1) m = fmaxf(m, __shfl_xor(m, off, 16));
      if (l16 == 0) red[wave][mt * 16 + hf * 8 + r] = m;
    }
  __syncthreads();
  if (tid < 32) {
    float m = -3.0e38f;
    #pragma unroll
    for (int w = 0; w < 8; ++w) m = fmaxf(m, red[w][tid]);
    stat[tid] = m;
  }
  __syncthreads();

  // --- p = exp(s - max) * mask ---
  #pragma unroll
  for (int mt = 0; mt < 2; ++mt)
    #pragma unroll
    for (int ni = 0; ni < 8; ++ni)
      #pragma unroll
      for (int r = 0; r < 8; ++r)
        acc[mt][ni][r] = __expf(acc[mt][ni][r] - stat[mt * 16 + hf * 8 + r]) * mk[ni];

  // --- row sum ---
  #pragma unroll
  for (int mt = 0; mt < 2; ++mt)
    #pragma unroll
    for (int r = 0; r < 8; ++r) {
      float s = 0.0f;
      #pragma unroll
      for (int ni = 0; ni < 8; ++ni) s += acc[mt][ni][r];
      #pragma unroll
      for (int off = 8; off >= 1; off >>= 1) s += __shfl_xor(s, off, 16);
      if (l16 == 0) red[wave][mt * 16 + hf * 8 + r] = s;
    }
  __syncthreads();
  if (tid < 32) {
    float s = 0.0f;
    #pragma unroll
    for (int w = 0; w < 8; ++w) s += red[w][tid];
    stat[tid] = s;
  }
  __syncthreads();

  // --- normalize + store bf16 ---
  #pragma unroll
  for (int mt = 0; mt < 2; ++mt)
    #pragma unroll
    for (int r = 0; r < 8; ++r) {
      const int rowid = mt * 16 + hf * 8 + r;
      const float inv = 1.0f / (stat[rowid] + 1e-13f);
      #pragma unroll
      for (int ni = 0; ni < 8; ++ni)
        probs[((size_t)b * kLq + q0 + rowid) * kLk + nb + ni * 16 + l16] =
            f2bf(acc[mt][ni][r] * inv);
    }
}

// ---------------------------------------------------------------------------
// Kernel 3: context = probs @ memory (NN GEMM, K = Lk).  Memory tile is
// converted fp32->bf16 and transposed into LDS (two K-rows packed per DWORD
// store).  Double-buffered.  Block tile 128x128, wave 64x32 -> out[:, :768].
// ---------------------------------------------------------------------------
__device__ __forceinline__ void load_mem_tileT(
    const float* __restrict__ Mp, unsigned short (*Ms)[40],
    int d0, int kb, int ldk2, int ldc)
{
  #pragma unroll
  for (int pp = 0; pp < 2; ++pp) {                // 2 passes x 8 k-pairs x 128 cols
    const int kk = pp * 16 + ldk2 * 2;
    float4 m0v = *(const float4*)(Mp + (size_t)(kb + kk) * kD + d0 + ldc);
    float4 m1v = *(const float4*)(Mp + (size_t)(kb + kk + 1) * kD + d0 + ldc);
    *(unsigned*)&Ms[ldc + 0][kk] = cvt_pk_bf16(m0v.x, m1v.x);
    *(unsigned*)&Ms[ldc + 1][kk] = cvt_pk_bf16(m0v.y, m1v.y);
    *(unsigned*)&Ms[ldc + 2][kk] = cvt_pk_bf16(m0v.z, m1v.z);
    *(unsigned*)&Ms[ldc + 3][kk] = cvt_pk_bf16(m0v.w, m1v.w);
  }
}

__global__ __launch_bounds__(256) void context_kernel(
    const unsigned short* __restrict__ probs,
    const float* __restrict__ mem,
    float* __restrict__ out)
{
  __shared__ unsigned short Ms[2][128][40];       // [d][k] transposed tile

  const int tid  = threadIdx.x;
  const int b    = blockIdx.z;
  const int q0   = blockIdx.x * 128;
  const int d0   = blockIdx.y * 128;
  const int wave = tid >> 5, lane = tid & 31;
  const int hf   = lane >> 4, l16 = lane & 15;
  const int wm   = wave & 1, wn = wave >> 1;
  const int ldk2 = tid >> 5;                      // 0..7 (k-pair index)
  const int ldc  = (tid & 31) * 4;                // 0..124 (d col, float4)

  const unsigned short* Pp = probs + ((size_t)b * kLq + q0) * kLk;
  const float* Mp = mem + (size_t)b * kLk * kD;

  v8f acc[4][2] = {};

  load_mem_tileT(Mp, Ms[0], d0, 0, ldk2, ldc);
  __syncthreads();

  for (int kt = 0; kt < kLk / 32; ++kt) {
    const int kb  = kt * 32;
    const int cur = kt & 1;
    if (kt + 1 < kLk / 32)                        // prefetch next tile
      load_mem_tileT(Mp, Ms[cur ^ 1], d0, kb + 32, ldk2, ldc);

    ABFrag a[4], bf2[2];
    #pragma unroll
    for (int mt = 0; mt < 4; ++mt) {
      const unsigned short* p = Pp + (size_t)(wm * 64 + mt * 16 + l16) * kLk + kb;
      a[mt].h[0] = *(const v8bf*)(p + hf * 8);
      a[mt].h[1] = *(const v8bf*)(p + 16 + hf * 8);
    }
    #pragma unroll
    for (int nt = 0; nt < 2; ++nt) {
      const int col = wn * 32 + nt * 16 + l16;
      bf2[nt].h[0] = *(const v8bf*)&Ms[cur][col][hf * 8];
      bf2[nt].h[1] = *(const v8bf*)&Ms[cur][col][16 + hf * 8];
    }
    #pragma unroll
    for (int mt = 0; mt < 4; ++mt)
      #pragma unroll
      for (int nt = 0; nt < 2; ++nt)
        acc[mt][nt] = __builtin_amdgcn_wmma_f32_16x16x32_bf16(
            false, a[mt].v, false, bf2[nt].v, (short)0, acc[mt][nt], false, false);

    __syncthreads();
  }

  #pragma unroll
  for (int mt = 0; mt < 4; ++mt)
    #pragma unroll
    for (int nt = 0; nt < 2; ++nt)
      #pragma unroll
      for (int r = 0; r < 8; ++r) {
        const int row = q0 + wm * 64 + mt * 16 + hf * 8 + r;
        const int col = d0 + wn * 32 + nt * 16 + l16;
        out[((size_t)b * kLq + row) * (2 * kD) + col] = acc[mt][nt][r];
      }
}

// ---------------------------------------------------------------------------
// Kernel 4: out[:, :, 768:1536] = inp  (vectorized copy)
// ---------------------------------------------------------------------------
__global__ __launch_bounds__(256) void concat_copy_kernel(
    const float* __restrict__ inp, float* __restrict__ out)
{
  const size_t i  = ((size_t)blockIdx.x * blockDim.x + threadIdx.x) * 4;
  const size_t bq = i / kD;
  const int    d  = (int)(i % kD);                // kD % 4 == 0: float4 stays in-row
  float4 v = *(const float4*)(inp + i);
  *(float4*)(out + bq * (2 * kD) + kD + d) = v;
}

// ---------------------------------------------------------------------------
extern "C" void kernel_launch(void* const* d_in, const int* in_sizes, int n_in,
                              void* d_out, int out_size, void* d_ws, size_t ws_size,
                              hipStream_t stream)
{
  const float* inp  = (const float*)d_in[0];   // [B, Lq, D]
  const float* memb = (const float*)d_in[1];   // [B, Lk, D]
  const int*   mask = (const int*)d_in[2];     // [B, Lk]
  const float* W    = (const float*)d_in[3];   // [A, D]
  float* out = (float*)d_out;                  // [B, Lq, 2D]

  unsigned short* ws    = (unsigned short*)d_ws;
  unsigned short* inp_p = ws;                                   // B*Lq*A bf16
  unsigned short* mem_p = inp_p + (size_t)kB * kLq * kA;        // B*Lk*A bf16
  unsigned short* probs = mem_p + (size_t)kB * kLk * kA;        // B*Lq*Lk bf16

  proj_relu_kernel<<<dim3((kB * kLq) / 128, kA / 128), 256, 0, stream>>>(
      inp, W, inp_p);
  proj_relu_kernel<<<dim3((kB * kLk) / 128, kA / 128), 256, 0, stream>>>(
      memb, W, mem_p);
  scores_softmax_kernel<<<dim3(kLq / 32, kB), 256, 0, stream>>>(
      inp_p, mem_p, mask, probs);
  context_kernel<<<dim3(kLq / 128, kD / 128, kB), 256, 0, stream>>>(
      probs, memb, out);
  concat_copy_kernel<<<(unsigned)((size_t)kB * kLq * kD / 4 / 256), 256, 0, stream>>>(
      inp, out);
}
